// EnhancedCEGNet_20246475833469
// MI455X (gfx1250) — compile-verified, matrix operand order checked by reading, not run on hardware
//
#include <hip/hip_runtime.h>
#include <hip/hip_bf16.h>
#include <math.h>

// ---------------------------------------------------------------------------
// Types for CDNA5 WMMA (wave32, 16x16x32 bf16 -> f32)
// ---------------------------------------------------------------------------
typedef __attribute__((ext_vector_type(16))) __bf16         v16bf;
typedef __attribute__((ext_vector_type(16))) unsigned short v16u;
typedef __attribute__((ext_vector_type(8)))  unsigned short u16x8;
typedef __attribute__((ext_vector_type(4)))  unsigned short u16x4;
typedef __attribute__((ext_vector_type(8)))  float          v8f;

#define NN   8192
#define EE   262144
#define HH   128
#define GG   16
#define NHEADS 2

__device__ __forceinline__ unsigned short f2bf(float f) {
  unsigned int u = __float_as_uint(f);
  unsigned int r = 0x7FFFu + ((u >> 16) & 1u);
  return (unsigned short)((u + r) >> 16);
}

__device__ __forceinline__ u16x4 pack4(float4 v) {
  u16x4 p = {f2bf(v.x), f2bf(v.y), f2bf(v.z), f2bf(v.w)};
  return p;
}

// Load one 16x32 bf16 A/B fragment for V_WMMA_*_16X16X32.
// Per ISA 7.12.2: lane l, element i -> K = (i&7) + 16*(i>>3) + 8*(l>=16).
// With the fragment's K-dim stored innermost and p pointing at K = 8*hi of the
// lane's row, elements 0..7 are p[0..7] and 8..15 are p[16..23]:
// exactly two 16-byte LDS loads (ds_load_b128).
__device__ __forceinline__ v16bf ldfrag(const unsigned short* p) {
  u16x8 lo = *(const u16x8*)p;
  u16x8 hi = *(const u16x8*)(p + 16);
  v16u u = __builtin_shufflevector(lo, hi, 0, 1, 2, 3, 4, 5, 6, 7,
                                   8, 9, 10, 11, 12, 13, 14, 15);
  return __builtin_bit_cast(v16bf, u);
}

__device__ __forceinline__ v8f wmma_bf16(v16bf a, v16bf b, v8f c) {
  return __builtin_amdgcn_wmma_f32_16x16x32_bf16(false, a, false, b,
                                                 (short)0, c, false, false);
}

// ---------------------------------------------------------------------------
// Tiled GEMM:  C[M,Nout] = act(A[M,K] @ B[K,Nout] + bias), K % 32 == 0.
// Software-pipelined: tile k+1 is fetched into registers (global_load_b128 x4
// in flight) while tile k is consumed by WMMA from LDS.
// 256 threads = 8 waves; block tile 64x64; each wave owns a 16x32 subtile.
// act: 0=none 1=relu 2=sigmoid
// ---------------------------------------------------------------------------
__global__ __launch_bounds__(256) void wmma_gemm_k(
    const float* __restrict__ A, const float* __restrict__ B,
    const float* __restrict__ bias, float* __restrict__ C,
    int M, int K, int Nout, int ldc, int act)
{
  __shared__ __align__(16) unsigned short As[64][40];   // [m][k], 80B stride
  __shared__ __align__(16) unsigned short Bst[64][40];  // [n][k] (transposed)
  const int t    = threadIdx.x;
  const int lane = t & 31;
  const int wave = t >> 5;
  const int m0   = blockIdx.x * 64;
  const int n0   = blockIdx.y * 64;
  const int mt   = wave & 3;   // M sub-tile 0..3
  const int nh   = wave >> 2;  // N half 0..1
  const int lrow = lane & 15;
  const int hi   = lane >> 4;

  // per-thread staging coordinates (A tile 64x32 = 512 float4, 2/thread;
  // B tile 32x64 = 512 float4, 2/thread)
  const int ia0 = t, ia1 = t + 256;
  const int ra0 = ia0 >> 3, ca0 = (ia0 & 7) << 2;
  const int ra1 = ia1 >> 3, ca1 = (ia1 & 7) << 2;
  const int rb0 = ia0 >> 4, cb0 = (ia0 & 15) << 2;
  const int rb1 = ia1 >> 4, cb1 = (ia1 & 15) << 2;

  v8f acc0 = {};
  v8f acc1 = {};

  // preload tile 0
  float4 a0 = *(const float4*)&A[(size_t)(m0 + ra0) * K + ca0];
  float4 a1 = *(const float4*)&A[(size_t)(m0 + ra1) * K + ca1];
  float4 b0 = *(const float4*)&B[(size_t)rb0 * Nout + n0 + cb0];
  float4 b1 = *(const float4*)&B[(size_t)rb1 * Nout + n0 + cb1];

  for (int k0 = 0; k0 < K; k0 += 32) {
    // commit held registers to LDS
    *(u16x4*)&As[ra0][ca0] = pack4(a0);
    *(u16x4*)&As[ra1][ca1] = pack4(a1);
    Bst[cb0 + 0][rb0] = f2bf(b0.x);
    Bst[cb0 + 1][rb0] = f2bf(b0.y);
    Bst[cb0 + 2][rb0] = f2bf(b0.z);
    Bst[cb0 + 3][rb0] = f2bf(b0.w);
    Bst[cb1 + 0][rb1] = f2bf(b1.x);
    Bst[cb1 + 1][rb1] = f2bf(b1.y);
    Bst[cb1 + 2][rb1] = f2bf(b1.z);
    Bst[cb1 + 3][rb1] = f2bf(b1.w);
    __syncthreads();

    // issue next-tile loads (clamped on last iteration -> unconditional)
    int kn = (k0 + 32 < K) ? k0 + 32 : k0;
    a0 = *(const float4*)&A[(size_t)(m0 + ra0) * K + kn + ca0];
    a1 = *(const float4*)&A[(size_t)(m0 + ra1) * K + kn + ca1];
    b0 = *(const float4*)&B[(size_t)(kn + rb0) * Nout + n0 + cb0];
    b1 = *(const float4*)&B[(size_t)(kn + rb1) * Nout + n0 + cb1];
    __builtin_amdgcn_sched_barrier(0);  // keep loads above the compute

    v16bf a  = ldfrag(&As[mt * 16 + lrow][hi << 3]);
    v16bf f0 = ldfrag(&Bst[nh * 32 + lrow][hi << 3]);
    v16bf f1 = ldfrag(&Bst[nh * 32 + 16 + lrow][hi << 3]);
    acc0 = wmma_bf16(a, f0, acc0);
    acc1 = wmma_bf16(a, f1, acc1);
    __syncthreads();
  }

#pragma unroll
  for (int r = 0; r < 8; ++r) {
    int m  = m0 + mt * 16 + r + (hi << 3);
    int c0 = n0 + nh * 32 + lrow;
    int c1 = c0 + 16;
    float v0 = acc0[r] + (bias ? bias[c0] : 0.0f);
    float v1 = acc1[r] + (bias ? bias[c1] : 0.0f);
    if (act == 1) { v0 = fmaxf(v0, 0.0f); v1 = fmaxf(v1, 0.0f); }
    else if (act == 2) {
      v0 = 1.0f / (1.0f + __expf(-v0));
      v1 = 1.0f / (1.0f + __expf(-v1));
    }
    if (m < M) {
      C[(size_t)m * ldc + c0] = v0;
      C[(size_t)m * ldc + c1] = v1;
    }
  }
}

// ---------------------------------------------------------------------------
// Flash attention: O = softmax(Q K^T / sqrt(H)) V, H=128.
// 128 threads = 4 waves; block handles 64 query rows (16 per wave).
// Keys in tiles of 32, software-pipelined: next K/V tile is fetched into
// registers (16x global_load_b128 in flight) while the current tile runs
// S = QK^T, online softmax, and O += PV on the WMMA pipe.
// Q/K tiles stored [row][feat]; V stored transposed [col][key] so every
// fragment reads contiguous K (2x ds_load_b128). P transpose goes through
// per-wave LDS, ordered with s_wait_dscnt (per-wave DS ops are in-order).
// ---------------------------------------------------------------------------
__global__ __launch_bounds__(128) void flash_attn_k(
    const float* __restrict__ Q, const float* __restrict__ Km,
    const float* __restrict__ V, float* __restrict__ O,
    int nkeys, int ldo, float qscale)
{
  __shared__ __align__(16) unsigned short Qs[64][136];   // 272B stride
  __shared__ __align__(16) unsigned short Ks[32][136];
  __shared__ __align__(16) unsigned short Vst[128][40];  // [vcol][key]
  __shared__ __align__(16) unsigned short Ps[4][16][40];

  const int t    = threadIdx.x;
  const int lane = t & 31;
  const int wave = t >> 5;
  const int lrow = lane & 15;
  const int hi   = lane >> 4;
  const int q0   = blockIdx.x * 64;

  // stage Q (64x128 = 2048 float4, 16/thread, groups of 4 loads in flight)
#pragma unroll
  for (int grp = 0; grp < 4; ++grp) {
    float4 v[4];
    int r[4], c4[4];
#pragma unroll
    for (int j = 0; j < 4; ++j) {
      int idx = t + (grp * 4 + j) * 128;
      r[j] = idx >> 5; c4[j] = (idx & 31) << 2;
      v[j] = *(const float4*)&Q[(size_t)(q0 + r[j]) * HH + c4[j]];
    }
#pragma unroll
    for (int j = 0; j < 4; ++j) {
      v[j].x *= qscale; v[j].y *= qscale; v[j].z *= qscale; v[j].w *= qscale;
      *(u16x4*)&Qs[r[j]][c4[j]] = pack4(v[j]);
    }
  }
  __syncthreads();

  v16bf qf[4];
#pragma unroll
  for (int kf = 0; kf < 4; ++kf)
    qf[kf] = ldfrag(&Qs[wave * 16 + lrow][kf * 32 + (hi << 3)]);

  // per-thread K/V staging coordinates (32x128 = 1024 float4, 8/thread each)
  int rr[8], cc[8];
#pragma unroll
  for (int q = 0; q < 8; ++q) {
    int idx = t + q * 128;
    rr[q] = idx >> 5;
    cc[q] = (idx & 31) << 2;
  }

  float4 kv[8], vv[8];
#pragma unroll
  for (int q = 0; q < 8; ++q)  // preload tile 0
    kv[q] = *(const float4*)&Km[(size_t)rr[q] * HH + cc[q]];
#pragma unroll
  for (int q = 0; q < 8; ++q)
    vv[q] = *(const float4*)&V[(size_t)rr[q] * HH + cc[q]];

  float mrow[8], lsum[8];
  v8f o[8];
  v8f zed = {};
#pragma unroll
  for (int r = 0; r < 8; ++r) { mrow[r] = -1e30f; lsum[r] = 0.0f; }
#pragma unroll
  for (int j = 0; j < 8; ++j) o[j] = zed;

  for (int kb = 0; kb < nkeys; kb += 32) {
    __syncthreads();  // previous-tile readers done
    // commit held registers to LDS
#pragma unroll
    for (int q = 0; q < 8; ++q) {
      *(u16x4*)&Ks[rr[q]][cc[q]] = pack4(kv[q]);
      Vst[cc[q] + 0][rr[q]] = f2bf(vv[q].x);
      Vst[cc[q] + 1][rr[q]] = f2bf(vv[q].y);
      Vst[cc[q] + 2][rr[q]] = f2bf(vv[q].z);
      Vst[cc[q] + 3][rr[q]] = f2bf(vv[q].w);
    }
    __syncthreads();

    // issue next-tile loads (clamped on last iteration -> unconditional)
    int kn = (kb + 32 < nkeys) ? kb + 32 : kb;
#pragma unroll
    for (int q = 0; q < 8; ++q)
      kv[q] = *(const float4*)&Km[(size_t)(kn + rr[q]) * HH + cc[q]];
#pragma unroll
    for (int q = 0; q < 8; ++q)
      vv[q] = *(const float4*)&V[(size_t)(kn + rr[q]) * HH + cc[q]];
    __builtin_amdgcn_sched_barrier(0);  // keep loads above the compute

    // S = Q @ K^T for 32 keys (two 16-key n-tiles)
    v8f s0 = zed, s1 = zed;
#pragma unroll
    for (int kf = 0; kf < 4; ++kf) {
      v16bf b0 = ldfrag(&Ks[lrow][kf * 32 + (hi << 3)]);
      v16bf b1 = ldfrag(&Ks[16 + lrow][kf * 32 + (hi << 3)]);
      s0 = wmma_bf16(qf[kf], b0, s0);
      s1 = wmma_bf16(qf[kf], b1, s1);
    }

    // online softmax; row m = r + 8*hi, spread over a 16-lane half
#pragma unroll
    for (int r = 0; r < 8; ++r) {
      float v = fmaxf(s0[r], s1[r]);
      v = fmaxf(v, __shfl_xor(v, 1));
      v = fmaxf(v, __shfl_xor(v, 2));
      v = fmaxf(v, __shfl_xor(v, 4));
      v = fmaxf(v, __shfl_xor(v, 8));
      float mnew  = fmaxf(mrow[r], v);
      float scale = __expf(mrow[r] - mnew);
      float p0 = __expf(s0[r] - mnew);
      float p1 = __expf(s1[r] - mnew);
      float rs = p0 + p1;
      rs += __shfl_xor(rs, 1);
      rs += __shfl_xor(rs, 2);
      rs += __shfl_xor(rs, 4);
      rs += __shfl_xor(rs, 8);
      lsum[r] = lsum[r] * scale + rs;
      mrow[r] = mnew;
      s0[r] = p0; s1[r] = p1;
#pragma unroll
      for (int j = 0; j < 8; ++j) o[j][r] *= scale;
    }

    // transpose P: D layout (lane=col) -> A layout (lane=row) via private LDS
#pragma unroll
    for (int r = 0; r < 8; ++r) {
      int m = r + (hi << 3);
      Ps[wave][m][lrow]      = f2bf(s0[r]);
      Ps[wave][m][16 + lrow] = f2bf(s1[r]);
    }
    asm volatile("s_wait_dscnt 0" ::: "memory");  // wave's DS writes complete
    v16bf pf = ldfrag(&Ps[wave][lrow][hi << 3]);

    // O += P @ V  (8 column tiles of V)
#pragma unroll
    for (int j = 0; j < 8; ++j) {
      v16bf bv = ldfrag(&Vst[j * 16 + lrow][hi << 3]);
      o[j] = wmma_bf16(pf, bv, o[j]);
    }
  }

#pragma unroll
  for (int j = 0; j < 8; ++j)
#pragma unroll
    for (int r = 0; r < 8; ++r) {
      int m = q0 + wave * 16 + r + (hi << 3);
      int c = j * 16 + lrow;
      O[(size_t)m * ldo + c] = o[j][r] / lsum[r];
    }
}

// ---------------------------------------------------------------------------
// Small helper kernels
// ---------------------------------------------------------------------------
__global__ void fill_f32_k(float* p, float v, size_t n) {
  size_t i = (size_t)blockIdx.x * blockDim.x + threadIdx.x;
  if (i < n) p[i] = v;
}

// zero-pad edge_attr [E,16] -> [E,32] so the edge GEMM uses the K%32==0 path
__global__ void pad_edge_k(const float* __restrict__ ea,
                           float* __restrict__ ea32) {
  size_t i = (size_t)blockIdx.x * blockDim.x + threadIdx.x;
  if (i < (size_t)EE * 32) {
    size_t e = i >> 5;
    int c = (int)(i & 31);
    ea32[i] = (c < 16) ? ea[e * 16 + c] : 0.0f;
  }
}

// Cw[gridDim.x][128] = A[I][J] @ B[J][128], zero rows past I (weight fusion)
__global__ __launch_bounds__(128) void fuse_w_k(const float* __restrict__ A,
                                                const float* __restrict__ B,
                                                float* __restrict__ Cw,
                                                int I, int J) {
  int i = blockIdx.x, n = threadIdx.x;
  float acc = 0.0f;
  if (i < I)
    for (int j = 0; j < J; ++j) acc += A[i * J + j] * B[j * HH + n];
  Cw[i * HH + n] = acc;
}

// bout[n] = bvec @ B + badd
__global__ __launch_bounds__(128) void fuse_b_k(const float* __restrict__ bvec,
                                                const float* __restrict__ B,
                                                const float* __restrict__ badd,
                                                float* __restrict__ bout, int J) {
  int n = threadIdx.x;
  float acc = badd ? badd[n] : 0.0f;
  for (int j = 0; j < J; ++j) acc += bvec[j] * B[j * HH + n];
  bout[n] = acc;
}

// aggr[dst[e]] += relu(nm[src[e]] + em[e])   one block per edge, 128 feats
__global__ __launch_bounds__(128) void scatter_msg_k(
    const float* __restrict__ nm, const float* __restrict__ em,
    const int* __restrict__ src, const int* __restrict__ dst,
    float* __restrict__ aggr) {
  int e = blockIdx.x, f = threadIdx.x;
  int s = src[e], d = dst[e];
  float v = nm[(size_t)s * HH + f] + em[(size_t)e * HH + f];
  v = fmaxf(v, 0.0f);
  atomicAdd(&aggr[(size_t)d * HH + f], v);
}

__global__ __launch_bounds__(256) void concat2_k(const float* __restrict__ a,
                                                 const float* __restrict__ b,
                                                 float* __restrict__ o) {
  int n = blockIdx.x, t = threadIdx.x;
  o[(size_t)n * 256 + t] =
      (t < 128) ? a[(size_t)n * 128 + t] : b[(size_t)n * 128 + (t - 128)];
}

__global__ void gated_k(const float* __restrict__ g, const float* __restrict__ u,
                        const float* __restrict__ oldx, float* __restrict__ out,
                        size_t n) {
  size_t i = (size_t)blockIdx.x * blockDim.x + threadIdx.x;
  if (i < n) {
    float gg = g[i];
    out[i] = gg * u[i] + (1.0f - gg) * oldx[i];
  }
}

__global__ void bn_relu_res_k(const float* __restrict__ x,
                              const float* __restrict__ beta,
                              const float* __restrict__ gamma,
                              const float* __restrict__ mean,
                              const float* __restrict__ var,
                              const float* __restrict__ res,
                              float* __restrict__ out, size_t n) {
  size_t i = (size_t)blockIdx.x * blockDim.x + threadIdx.x;
  if (i < n) {
    int f = (int)(i & 127);
    float v = (x[i] - mean[f]) * rsqrtf(var[f] + 1e-5f) * gamma[f] + beta[f];
    v = fmaxf(v, 0.0f);
    if (res) v += res[i];
    out[i] = v;
  }
}

// pooling: sum / count / bitwise max (x3 >= 0 by construction)
__global__ __launch_bounds__(128) void pool_k(const float* __restrict__ x,
                                              const int* __restrict__ batch,
                                              float* __restrict__ sum,
                                              unsigned int* __restrict__ mx,
                                              float* __restrict__ cnt) {
  int n = blockIdx.x, f = threadIdx.x;
  int g = batch[n];
  float v = x[(size_t)n * HH + f];
  atomicAdd(&sum[g * HH + f], v);
  atomicMax(&mx[g * HH + f], __float_as_uint(v));
  if (f == 0) atomicAdd(&cnt[g], 1.0f);
}

// final per-graph MLP: [mean|max|sum](384) -> 128 -> 64 -> 1
__global__ __launch_bounds__(128) void head_mlp_k(
    const float* __restrict__ sum, const unsigned int* __restrict__ mx,
    const float* __restrict__ cnt, const float* __restrict__ fc1w,
    const float* __restrict__ fc1b, const float* __restrict__ fc2w,
    const float* __restrict__ fc2b, const float* __restrict__ rw,
    const float* __restrict__ rb, float* __restrict__ out) {
  __shared__ float comb[384];
  __shared__ float h1[128];
  __shared__ float h2[64];
  int g = blockIdx.x, t = threadIdx.x;
  float c = fmaxf(cnt[g], 1.0f);
  float s = sum[g * HH + t];
  comb[t]       = s / c;
  comb[128 + t] = __uint_as_float(mx[g * HH + t]);
  comb[256 + t] = s;
  __syncthreads();
  float a = fc1b[t];
  for (int j = 0; j < 384; ++j) a += comb[j] * fc1w[j * 128 + t];
  h1[t] = fmaxf(a, 0.0f);
  __syncthreads();
  if (t < 64) {
    float b = fc2b[t];
    for (int j = 0; j < 128; ++j) b += h1[j] * fc2w[j * 64 + t];
    h2[t] = fmaxf(b, 0.0f);
  }
  __syncthreads();
  if (t == 0) {
    float b = rb[0];
    for (int j = 0; j < 64; ++j) b += h2[j] * rw[j];
    out[g] = b;
  }
}

// ---------------------------------------------------------------------------
// Orchestration
// ---------------------------------------------------------------------------
extern "C" void kernel_launch(void* const* d_in, const int* in_sizes, int n_in,
                              void* d_out, int out_size, void* d_ws,
                              size_t ws_size, hipStream_t stream) {
  (void)in_sizes; (void)n_in; (void)out_size; (void)ws_size;
  const float* x_in  = (const float*)d_in[0];
  const int*   eidx  = (const int*)d_in[1];
  const int*   src   = eidx;
  const int*   dst   = eidx + EE;
  const float* eattr = (const float*)d_in[2];
  const int*   batch = (const int*)d_in[3];
  auto P = [&](int i) { return (const float*)d_in[i]; };

  // param leaf indices (pytree: dicts flattened in sorted-key order)
  const int bnBase[3]   = {4, 8, 12};      // beta,gamma,mean,var
  const int convBase[3] = {16, 42, 68};
  // conv leaves: attn[h]{k.b,k.w,q.b,q.w,v.b,v.w} (h*6), then:
  const int C_COMB_B = 12, C_COMB_W = 13, C_EDGE_B = 14, C_EDGE_W = 15;
  const int C_GATE_B = 16, C_GATE_W = 17, C_MSG_B = 18, C_MSG_W = 19;
  const int C_SND_B  = 20, C_SND_W  = 21, C_UPD_B = 22, C_UPD_W = 23;
  const int C_UX_B   = 24, C_UX_W   = 25;
  const int FC1_B = 94, FC1_W = 95, FC2_B = 96, FC2_W = 97, REG_B = 98, REG_W = 99;

  // workspace allocator
  char* base = (char*)d_ws;
  size_t off = 0;
  auto alloc = [&](size_t bytes) -> float* {
    float* p = (float*)(base + off);
    off = (off + bytes + 255) & ~(size_t)255;
    return p;
  };
  float* em       = alloc((size_t)EE * HH * 4);  // 134 MB, reused per layer
  float* ea32     = alloc((size_t)EE * 32 * 4);  // zero-padded edge_attr
  float* nm       = alloc((size_t)NN * HH * 4);
  float* aggr     = alloc((size_t)NN * HH * 4);
  float* oldx     = alloc((size_t)NN * HH * 4);
  float* qb       = alloc((size_t)NN * HH * 4);
  float* kb       = alloc((size_t)NN * HH * 4);
  float* vb       = alloc((size_t)NN * HH * 4);
  float* head_out = alloc((size_t)NN * 2 * HH * 4);
  float* aggr2    = alloc((size_t)NN * HH * 4);
  float* catb     = alloc((size_t)NN * 2 * HH * 4);
  float* gate     = alloc((size_t)NN * HH * 4);
  float* upd      = alloc((size_t)NN * HH * 4);
  float* xpre     = alloc((size_t)NN * HH * 4);
  float* xl[3]    = {alloc((size_t)NN * HH * 4), alloc((size_t)NN * HH * 4),
                     alloc((size_t)NN * HH * 4)};
  float* Wsf  = alloc(128 * 128 * 4);
  float* bsf  = alloc(128 * 4);
  float* Wf   = alloc(32 * 128 * 4);   // 32 rows: 16 fused + 16 zero
  float* bf   = alloc(128 * 4);
  float* psum = alloc(GG * HH * 4);
  float* pmx  = alloc(GG * HH * 4);
  float* pcnt = alloc(GG * 4);

  const float qscale = 1.0f / sqrtf((float)HH);
  auto gemm = [&](const float* A, const float* B, const float* bias, float* C,
                  int M, int K, int ldc, int act) {
    wmma_gemm_k<<<dim3(M / 64, HH / 64), 256, 0, stream>>>(A, B, bias, C, M, K,
                                                           HH, ldc, act);
  };

  // one-time: zero-padded edge features (K=16 -> K=32)
  pad_edge_k<<<(EE * 32 + 255) / 256, 256, 0, stream>>>(eattr, ea32);

  const float* xin = x_in;
  int nin = 64;
  const float* resid = nullptr;

  for (int L = 0; L < 3; ++L) {
    int cb = convBase[L], bb = bnBase[L];
    const float* msgW = P(cb + C_MSG_W);  // [256,128]: top=node, bottom=edge
    // fuse weights: nm = x @ (Wsnd @ Wmsg_top) + (bsnd @ Wmsg_top)
    fuse_w_k<<<nin, 128, 0, stream>>>(P(cb + C_SND_W), msgW, Wsf, nin, 128);
    fuse_b_k<<<1, 128, 0, stream>>>(P(cb + C_SND_B), msgW, nullptr, bsf, 128);
    //              em = ea @ (Wedge @ Wmsg_bot) + (bedge @ Wmsg_bot + bmsg)
    fuse_w_k<<<32, 128, 0, stream>>>(P(cb + C_EDGE_W), msgW + 128 * 128, Wf,
                                     16, 128);
    fuse_b_k<<<1, 128, 0, stream>>>(P(cb + C_EDGE_B), msgW + 128 * 128,
                                    P(cb + C_MSG_B), bf, 128);

    gemm(xin, Wsf, bsf, nm, NN, nin, HH, 0);
    gemm(ea32, Wf, bf, em, EE, 32, HH, 0);

    fill_f32_k<<<(NN * HH + 255) / 256, 256, 0, stream>>>(aggr, 0.0f,
                                                          (size_t)NN * HH);
    scatter_msg_k<<<EE, 128, 0, stream>>>(nm, em, src, dst, aggr);

    gemm(xin, P(cb + C_UX_W), P(cb + C_UX_B), oldx, NN, nin, HH, 0);

    for (int h = 0; h < NHEADS; ++h) {
      int ab = cb + 6 * h;  // k.b,k.w,q.b,q.w,v.b,v.w
      gemm(aggr, P(ab + 3), P(ab + 2), qb, NN, HH, HH, 0);
      gemm(aggr, P(ab + 1), P(ab + 0), kb, NN, HH, HH, 0);
      gemm(aggr, P(ab + 5), P(ab + 4), vb, NN, HH, HH, 0);
      flash_attn_k<<<NN / 64, 128, 0, stream>>>(qb, kb, vb, head_out + h * HH,
                                                NN, 2 * HH, qscale);
    }
    gemm(head_out, P(cb + C_COMB_W), P(cb + C_COMB_B), aggr2, NN, 2 * HH, HH, 0);

    concat2_k<<<NN, 256, 0, stream>>>(oldx, aggr2, catb);
    gemm(catb, P(cb + C_GATE_W), P(cb + C_GATE_B), gate, NN, 2 * HH, HH, 2);
    gemm(catb, P(cb + C_UPD_W),  P(cb + C_UPD_B),  upd,  NN, 2 * HH, HH, 1);
    gated_k<<<(NN * HH + 255) / 256, 256, 0, stream>>>(gate, upd, oldx, xpre,
                                                       (size_t)NN * HH);
    bn_relu_res_k<<<(NN * HH + 255) / 256, 256, 0, stream>>>(
        xpre, P(bb + 0), P(bb + 1), P(bb + 2), P(bb + 3), resid, xl[L],
        (size_t)NN * HH);
    xin = xl[L];
    resid = xl[L];
    nin = HH;
  }

  // global pooling + readout MLP
  fill_f32_k<<<(2 * GG * HH + GG + 255) / 256, 256, 0, stream>>>(
      psum, 0.0f, (size_t)(2 * GG * HH + GG));  // psum, pmx, pcnt contiguous
  pool_k<<<NN, 128, 0, stream>>>(xl[2], batch, psum, (unsigned int*)pmx, pcnt);
  head_mlp_k<<<GG, 128, 0, stream>>>(psum, (unsigned int*)pmx, pcnt, P(FC1_W),
                                     P(FC1_B), P(FC2_W), P(FC2_B), P(REG_W),
                                     P(REG_B), (float*)d_out);
}
